// STDPNet_2336462209360
// MI455X (gfx1250) — matrix operand order repaired and checked by reading, not compiled, not run on hardware
//
#include <hip/hip_runtime.h>

// ---------------------------------------------------------------------------
// SNN forward (2-layer IF network) for MI455X / gfx1250, wave32 + WMMA bf16.
//   per step: v1 += x_t @ W1^T ; s1 = (v1>=1); v1 *= (1-s1)
//             v2 += s1  @ W2^T ; s2 = (v2>=1); v2 *= (1-s2); rec += s2
// GEMM1: 128x128 tile, double-buffered LDS fed by global_load_async_to_lds_b128
// (ASYNCcnt pipeline), v_wmma_f32_16x16x32_bf16 with fp32 accumulate.
// x/W pre-converted to bf16 once, K padded 784->800 so the hot loop has no
// guards and no conversions. s1 is {0,1} -> exact in bf16.
// ---------------------------------------------------------------------------

typedef __attribute__((ext_vector_type(16))) __bf16 v16bf;
typedef __attribute__((ext_vector_type(8)))  float  v8f;

#define T_STEPS 32
#define BATCH   1024
#define IN_DIM  784
#define K_PAD   800      // 784 padded to multiple of BK=32 (zeros)
#define HID     4096
#define OUT_DIM 10
#define OUT_PAD 16       // pad layer-2 N to one WMMA tile
#define NT      (K_PAD / 32)   // 25 K-tiles

union FragBF { v16bf v; uint4 u[2]; };

// 16B global -> LDS async copy (CDNA5, tracked by ASYNCcnt)
__device__ __forceinline__ void async_copy16(unsigned lds_off, const __bf16* gptr) {
  asm volatile("global_load_async_to_lds_b128 %0, %1, off"
               :: "v"(lds_off), "v"(gptr)
               : "memory");
}
#define WAIT_ASYNC(imm) asm volatile("s_wait_asynccnt " #imm ::: "memory")

// --- prep 1: one row per block; bf16-convert + K-pad x_seq and W1 -----------
__global__ __launch_bounds__(256)
void snn_prep_rows(const float* __restrict__ x, const float* __restrict__ W1,
                   __bf16* __restrict__ xbf, __bf16* __restrict__ W1bf) {
  int row = blockIdx.x;
  const float* src;
  __bf16* dst;
  if (row < T_STEPS * BATCH) {
    src = x + (size_t)row * IN_DIM;
    dst = xbf + (size_t)row * K_PAD;
  } else {
    row -= T_STEPS * BATCH;
    src = W1 + (size_t)row * IN_DIM;
    dst = W1bf + (size_t)row * K_PAD;
  }
  for (int k = threadIdx.x; k < K_PAD; k += 256)
    dst[k] = (k < IN_DIM) ? (__bf16)src[k] : (__bf16)0.0f;
}

// --- prep 2: W2 -> bf16 padded to 16 output rows ----------------------------
__global__ void snn_prep_w2(const float* __restrict__ W2, __bf16* __restrict__ W2bf) {
  int i = blockIdx.x * blockDim.x + threadIdx.x;
  if (i < OUT_PAD * HID) {
    int n = i >> 12;            // / HID
    int k = i & (HID - 1);
    W2bf[i] = (n < OUT_DIM) ? (__bf16)W2[n * HID + k] : (__bf16)0.0f;
  }
}

// --- layer 1: v1 += x_t @ W1^T, spike, hard reset, emit s1 (bf16) -----------
// block = 128(M) x 128(N) tile, 256 threads = 8 waves (2 x 4), BK = 32.
// A/B tiles staged by async global->LDS DMA, double-buffered.
__global__ __launch_bounds__(256)
void snn_layer1(const __bf16* __restrict__ xbf_t, const __bf16* __restrict__ W1bf,
                float* __restrict__ v1, __bf16* __restrict__ s1) {
  __shared__ __bf16 As[2][128 * 32];   // x tile  [m][k]
  __shared__ __bf16 Bs[2][128 * 32];   // W1 tile [n][k]

  const int tid  = threadIdx.x;
  const int lane = tid & 31;
  const int wave = tid >> 5;
  const int wm   = wave & 1;      // 64-row slab
  const int wn   = wave >> 1;     // 32-col slab
  const int m0   = blockIdx.y * 128;
  const int n0   = blockIdx.x * 128;
  const int mrow = lane & 15;
  const int hi   = lane >> 4;

  const __bf16* aBase = xbf_t + (size_t)m0 * K_PAD;   // + r*K_PAD + k
  const __bf16* bBase = W1bf  + (size_t)n0 * K_PAD;

  const unsigned asOff0 = (unsigned)(uintptr_t)&As[0][0];
  const unsigned asOff1 = (unsigned)(uintptr_t)&As[1][0];
  const unsigned bsOff0 = (unsigned)(uintptr_t)&Bs[0][0];
  const unsigned bsOff1 = (unsigned)(uintptr_t)&Bs[1][0];

  // per-thread chunk assignment: 2 A chunks + 2 B chunks of 16B per tile
  const int c0 = tid, c1 = tid + 256;            // chunk ids in [0,512)
  const int r0 = c0 >> 2, k0c = (c0 & 3) * 8;    // row, bf16 col within 32
  const int r1 = c1 >> 2, k1c = (c1 & 3) * 8;
  const unsigned l0 = (unsigned)(r0 * 32 + k0c) * 2;  // LDS byte offsets
  const unsigned l1 = (unsigned)(r1 * 32 + k1c) * 2;

  const v8f vzero = {0.f, 0.f, 0.f, 0.f, 0.f, 0.f, 0.f, 0.f};
  v8f acc[4][2];
  for (int a = 0; a < 4; ++a)
    for (int b = 0; b < 2; ++b) acc[a][b] = vzero;

  // prologue: stage tile 0 into buffer 0
  async_copy16(asOff0 + l0, aBase + (size_t)r0 * K_PAD + k0c);
  async_copy16(asOff0 + l1, aBase + (size_t)r1 * K_PAD + k1c);
  async_copy16(bsOff0 + l0, bBase + (size_t)r0 * K_PAD + k0c);
  async_copy16(bsOff0 + l1, bBase + (size_t)r1 * K_PAD + k1c);

  for (int kt = 0; kt < NT; ++kt) {
    const int cur = kt & 1;
    if (kt + 1 < NT) {
      // prefetch next tile into the other buffer (safe: it was last read in
      // iteration kt-1, which ended with a barrier). Its DMA latency overlaps
      // this iteration's compute; we only wait for tile kt (the 4 oldest).
      const int kn = (kt + 1) * 32;
      const unsigned an = cur ? asOff0 : asOff1;
      const unsigned bn = cur ? bsOff0 : bsOff1;
      async_copy16(an + l0, aBase + (size_t)r0 * K_PAD + kn + k0c);
      async_copy16(an + l1, aBase + (size_t)r1 * K_PAD + kn + k1c);
      async_copy16(bn + l0, bBase + (size_t)r0 * K_PAD + kn + k0c);
      async_copy16(bn + l1, bBase + (size_t)r1 * K_PAD + kn + k1c);
      WAIT_ASYNC(0x4);   // in-order ASYNCcnt: the 4 oldest (= tile kt) landed
    } else {
      WAIT_ASYNC(0x0);
    }
    __syncthreads();

    // load ALL fragments first (12 ds_load_b128), then 8 dense WMMAs; the
    // in-order DScnt lets the scheduler overlap loads with early WMMAs.
    FragBF fb[2];
    #pragma unroll
    for (int nt = 0; nt < 2; ++nt) {
      const __bf16* bb = &Bs[cur][(wn * 32 + nt * 16 + mrow) * 32 + 8 * hi];
      fb[nt].u[0] = *(const uint4*)(bb);
      fb[nt].u[1] = *(const uint4*)(bb + 16);
    }
    FragBF fa[4];
    #pragma unroll
    for (int mt = 0; mt < 4; ++mt) {
      const __bf16* ab = &As[cur][(wm * 64 + mt * 16 + mrow) * 32 + 8 * hi];
      fa[mt].u[0] = *(const uint4*)(ab);
      fa[mt].u[1] = *(const uint4*)(ab + 16);
    }
    #pragma unroll
    for (int mt = 0; mt < 4; ++mt)
      #pragma unroll
      for (int nt = 0; nt < 2; ++nt)
        acc[mt][nt] = __builtin_amdgcn_wmma_f32_16x16x32_bf16(
            false, fa[mt].v, false, fb[nt].v, (short)0, acc[mt][nt], false, false);
    __syncthreads();   // everyone done reading buf[cur] before it is re-filled
  }

  // fused IF-neuron epilogue (C layout: VGPR r -> M = r + 8*hi, N = lane&15)
  for (int mt = 0; mt < 4; ++mt) {
    int gr = m0 + wm * 64 + mt * 16 + hi * 8;
    for (int nt = 0; nt < 2; ++nt) {
      int gc = n0 + wn * 32 + nt * 16 + mrow;
      for (int r = 0; r < 8; ++r) {
        size_t idx = (size_t)(gr + r) * HID + gc;
        float v = v1[idx] + acc[mt][nt][r];
        float s = (v >= 1.0f) ? 1.0f : 0.0f;
        v1[idx] = v * (1.0f - s);
        s1[idx] = (__bf16)s;      // exact in bf16
      }
    }
  }
}

// --- layer 2: v2 += s1 @ W2^T (N padded to 16), spike, reset, rec += s2 -----
// 8 blocks x 256 threads; each wave owns a 16-row strip. K=4096 is split over
// 4 independent accumulators to break the serial WMMA dependency chain
// (XDL latency), then combined in a fixed deterministic order.
__global__ __launch_bounds__(256)
void snn_layer2(const __bf16* __restrict__ s1, const __bf16* __restrict__ W2bf,
                float* __restrict__ v2, float* __restrict__ rec) {
  const int tid  = threadIdx.x;
  const int lane = tid & 31;
  const int wave = tid >> 5;
  const int mrow = lane & 15;
  const int hi   = lane >> 4;
  const int row_base = blockIdx.x * 128 + wave * 16;

  const v8f vzero = {0.f, 0.f, 0.f, 0.f, 0.f, 0.f, 0.f, 0.f};
  v8f acc[4] = {vzero, vzero, vzero, vzero};

  const __bf16* arow = s1   + (size_t)(row_base + mrow) * HID;
  const __bf16* brow = W2bf + (size_t)mrow * HID + 8 * hi;

  for (int kk = 0; kk < HID; kk += 128) {      // 32 outer iterations
    #pragma unroll
    for (int j = 0; j < 4; ++j) {              // 4 independent K-streams
      const int k = kk + j * 32;
      FragBF fa, fb;
      fa.u[0] = *(const uint4*)(arow + k + 8 * hi);
      fa.u[1] = *(const uint4*)(arow + k + 16 + 8 * hi);
      fb.u[0] = *(const uint4*)(brow + k);
      fb.u[1] = *(const uint4*)(brow + k + 16);
      acc[j] = __builtin_amdgcn_wmma_f32_16x16x32_bf16(
          false, fa.v, false, fb.v, (short)0, acc[j], false, false);
    }
  }
  v8f accT = (acc[0] + acc[1]) + (acc[2] + acc[3]);

  const int row = row_base + hi * 8;
  for (int r = 0; r < 8; ++r) {
    int idx = (row + r) * OUT_PAD + mrow;
    float v = v2[idx] + accT[r];
    float s = (v >= 1.0f) ? 1.0f : 0.0f;
    v2[idx]  = v * (1.0f - s);
    rec[idx] += s;
  }
}

// --- strip the N padding into d_out [1024,10] -------------------------------
__global__ void snn_out(const float* __restrict__ rec, float* __restrict__ out) {
  int i = blockIdx.x * blockDim.x + threadIdx.x;
  if (i < BATCH * OUT_DIM) {
    int b = i / OUT_DIM;
    int o = i - b * OUT_DIM;
    out[i] = rec[b * OUT_PAD + o];
  }
}

// --- workspace layout (bytes) ----------------------------------------------
#define WS_V1    ((size_t)0)                    // 1024*4096*4      = 16777216
#define WS_S1    ((size_t)16777216)             // 1024*4096*2      =  8388608
#define WS_V2    ((size_t)25165824)             // 1024*16*4        =    65536
#define WS_REC   ((size_t)25231360)             // 1024*16*4        =    65536
#define WS_W2BF  ((size_t)25296896)             // 16*4096*2        =   131072
#define WS_W1BF  ((size_t)25427968)             // 4096*800*2       =  6553600
#define WS_XBF   ((size_t)31981568)             // 32*1024*800*2    = 52428800
#define WS_NEED  ((size_t)84410368)

extern "C" void kernel_launch(void* const* d_in, const int* in_sizes, int n_in,
                              void* d_out, int out_size, void* d_ws, size_t ws_size,
                              hipStream_t stream) {
  (void)in_sizes; (void)n_in; (void)out_size;
  if (ws_size < WS_NEED) return;  // need ~80.5 MiB of scratch

  const float* x_seq = (const float*)d_in[0];   // [32,1024,784] f32
  const float* W1    = (const float*)d_in[1];   // [4096,784]    f32
  const float* W2    = (const float*)d_in[2];   // [10,4096]     f32

  char*   ws   = (char*)d_ws;
  float*  v1   = (float*)(ws + WS_V1);
  __bf16* s1   = (__bf16*)(ws + WS_S1);
  float*  v2   = (float*)(ws + WS_V2);
  float*  rec  = (float*)(ws + WS_REC);
  __bf16* W2bf = (__bf16*)(ws + WS_W2BF);
  __bf16* W1bf = (__bf16*)(ws + WS_W1BF);
  __bf16* xbf  = (__bf16*)(ws + WS_XBF);

  // deterministic zero-init every call (graph-capture safe)
  hipMemsetAsync(v1, 0, (size_t)BATCH * HID * sizeof(float), stream);
  hipMemsetAsync(v2, 0, (size_t)BATCH * OUT_PAD * sizeof(float) * 2, stream); // v2+rec

  snn_prep_rows<<<T_STEPS * BATCH + HID, 256, 0, stream>>>(x_seq, W1, xbf, W1bf);
  snn_prep_w2<<<(OUT_PAD * HID + 255) / 256, 256, 0, stream>>>(W2, W2bf);

  dim3 g1(HID / 128, BATCH / 128);   // 32 x 8 = 256 blocks
  for (int t = 0; t < T_STEPS; ++t) {
    snn_layer1<<<g1, 256, 0, stream>>>(xbf + (size_t)t * BATCH * K_PAD, W1bf, v1, s1);
    snn_layer2<<<BATCH / 128, 256, 0, stream>>>(s1, W2bf, v2, rec);
  }

  snn_out<<<(BATCH * OUT_DIM + 255) / 256, 256, 0, stream>>>(rec, (float*)d_out);
}
